// Clusterer_21887153341082
// MI455X (gfx1250) — matrix operand order repaired and checked by reading, not compiled, not run on hardware
//
#include <hip/hip_runtime.h>
#include <hip/hip_bf16.h>

// ---------------------------------------------------------------------------
// Clusterer: out[n,k] = q/rowsum(q), q = 1/(1 + ||x_n - c_k||^2), ALPHA=1.
// N=131072, K=256, D=256 f32.
// Cross term x.c^T via v_wmma_f32_16x16x32_bf16 with hi/lo bf16 split
// (truncated hi + residual lo: 3 wmma per tile ~ f32 accuracy).
// Centers staged in LDS as bf16 hi/lo (padded stride, conflict-free b128).
// kt loop rolled; sched_barriers bound B-fragment liveness.
// Epilogue uses v_rcp_f32 + fma to minimize tail VALU.
// ---------------------------------------------------------------------------

typedef __attribute__((ext_vector_type(16))) __bf16 v16bf;
typedef __attribute__((ext_vector_type(8)))  float  v8f;

#define C_STRIDE 264   // 256 bf16 + 8 pad -> 528B row stride, conflict-free ds_load_b128

__device__ __forceinline__ float u32_as_f32(unsigned int u) {
    union { unsigned int u; float f; } v; v.u = u;
    return v.f;
}
__device__ __forceinline__ unsigned int f32_as_u32(float f) {
    union { float f; unsigned int u; } v; v.f = f;
    return v.u;
}

// pack truncated-bf16(a) into [15:0] and truncated-bf16(b) into [31:16]
__device__ __forceinline__ unsigned int pack2bf16_trunc(float a, float b) {
    return __builtin_amdgcn_perm(f32_as_u32(b), f32_as_u32(a), 0x07060302u);
}
// hi/lo split of a pair: h = packed hi (truncated), l = packed residual
__device__ __forceinline__ void split2bf16(float a, float b,
                                           unsigned int& h, unsigned int& l) {
    h = pack2bf16_trunc(a, b);
    float fa = u32_as_f32(f32_as_u32(a) & 0xffff0000u);
    float fb = u32_as_f32(f32_as_u32(b) & 0xffff0000u);
    l = pack2bf16_trunc(a - fa, b - fb);
}

__device__ __forceinline__ void unpack2bf16(unsigned int h, float& fa, float& fb) {
    fa = u32_as_f32(h << 16);
    fb = u32_as_f32(h & 0xffff0000u);
}

__device__ __forceinline__ float fast_rcp(float x) {
#if __has_builtin(__builtin_amdgcn_rcpf)
    return __builtin_amdgcn_rcpf(x);   // v_rcp_f32, ~1 ulp, TRANS pipe
#else
    return 1.f / x;
#endif
}

__device__ __forceinline__ v16bf load_b_frag(const unsigned short* lds, int elemIdx) {
    union { v16bf v; uint4 q[2]; } u;
    const uint4* p = reinterpret_cast<const uint4*>(lds + elemIdx);
    u.q[0] = p[0];
    u.q[1] = p[2];
    return u.v;
}

__device__ __forceinline__ void sched_fence() {
#if __has_builtin(__builtin_amdgcn_sched_barrier)
    // 0x76 = VALU|SALU|VMEM(all/read/write) may cross; WMMA + DS pinned.
    __builtin_amdgcn_sched_barrier(0x76);
#endif
}

__global__ __launch_bounds__(256, 1)
void clusterer_kernel(const float* __restrict__ x,
                      const float* __restrict__ cc,
                      float* __restrict__ out) {
    __shared__ __align__(16) unsigned short sc_hi[256 * C_STRIDE]; // 132 KB
    __shared__ __align__(16) unsigned short sc_lo[256 * C_STRIDE]; // 132 KB
    __shared__ float s_csq[256];
    __shared__ float s_xsq[8 * 16];

    const int tid  = threadIdx.x;
    const int wave = tid >> 5;
    const int lane = tid & 31;
    const int l15  = lane & 15;
    const int half = lane >> 4;

    // ---- Stage centers f32 -> bf16 hi/lo into LDS (coalesced, 8 f32/thread/iter)
    #pragma unroll 4
    for (int j = 0; j < 32; ++j) {
        int flat = j * 2048 + tid * 8;
        int row = flat >> 8;
        int d   = flat & 255;
        float4 f0 = *reinterpret_cast<const float4*>(cc + flat);
        float4 f1 = *reinterpret_cast<const float4*>(cc + flat + 4);
        float vals[8] = { f0.x, f0.y, f0.z, f0.w, f1.x, f1.y, f1.z, f1.w };
        union { unsigned int u[4]; uint4 q; } hi, lo;
        #pragma unroll
        for (int i = 0; i < 4; ++i) {
            split2bf16(vals[2 * i], vals[2 * i + 1], hi.u[i], lo.u[i]);
        }
        *reinterpret_cast<uint4*>(sc_hi + row * C_STRIDE + d) = hi.q;
        *reinterpret_cast<uint4*>(sc_lo + row * C_STRIDE + d) = lo.q;
    }
    __syncthreads();

    // ---- c_sq from staged (hi+lo): thread t owns center row t, uint4 sweeps
    {
        float cs = 0.f;
        const uint4* rh = reinterpret_cast<const uint4*>(sc_hi + tid * C_STRIDE);
        const uint4* rl = reinterpret_cast<const uint4*>(sc_lo + tid * C_STRIDE);
        #pragma unroll 4
        for (int d8 = 0; d8 < 32; ++d8) {
            uint4 hv = rh[d8];
            uint4 lv = rl[d8];
            const unsigned int hw[4] = { hv.x, hv.y, hv.z, hv.w };
            const unsigned int lw[4] = { lv.x, lv.y, lv.z, lv.w };
            #pragma unroll
            for (int i = 0; i < 4; ++i) {
                float ha, hb, la, lb;
                unpack2bf16(hw[i], ha, hb);
                unpack2bf16(lw[i], la, lb);
                float va = ha + la, vb = hb + lb;
                cs += va * va + vb * vb;
            }
        }
        s_csq[tid] = cs;
    }

    // ---- main compute: kt rolled (stream A), 16 col-tile accumulators live
    const int growBase = blockIdx.x * 128 + wave * 16;
    const float* xrow = x + (growBase + l15) * 256;
    const int bBase = l15 * C_STRIDE + half * 8;   // per-lane B base (elements)

    v8f acc[16];
    #pragma unroll
    for (int nt = 0; nt < 16; ++nt) acc[nt] = (v8f){0.f,0.f,0.f,0.f,0.f,0.f,0.f,0.f};

    float xsqp = 0.f;
    #pragma unroll 1
    for (int kt = 0; kt < 8; ++kt) {
        // load 16 f32 of x for this ktile, split into packed bf16 hi/lo
        const float* p = xrow + kt * 32 + half * 8;
        float4 g0 = *reinterpret_cast<const float4*>(p);
        float4 g1 = *reinterpret_cast<const float4*>(p + 4);
        float4 g2 = *reinterpret_cast<const float4*>(p + 16);
        float4 g3 = *reinterpret_cast<const float4*>(p + 20);
        float v0[16] = { g0.x, g0.y, g0.z, g0.w, g1.x, g1.y, g1.z, g1.w,
                         g2.x, g2.y, g2.z, g2.w, g3.x, g3.y, g3.z, g3.w };
        union { v16bf v; unsigned int u[8]; } uh, ul;
        #pragma unroll
        for (int i = 0; i < 8; ++i) {
            float a = v0[2 * i], b = v0[2 * i + 1];
            split2bf16(a, b, uh.u[i], ul.u[i]);
            xsqp += a * a + b * b;
        }
        const v16bf ah = uh.v;
        const v16bf al = ul.v;

        const int kOff = bBase + kt * 32;
        #pragma unroll
        for (int nt = 0; nt < 16; ++nt) {
            const int idx = kOff + nt * 16 * C_STRIDE;
            v16bf bh = load_b_frag(sc_hi, idx);
            v16bf bl = load_b_frag(sc_lo, idx);
            acc[nt] = __builtin_amdgcn_wmma_f32_16x16x32_bf16(false, ah, false, bh,
                                                              (short)0, acc[nt], false, false);
            acc[nt] = __builtin_amdgcn_wmma_f32_16x16x32_bf16(false, ah, false, bl,
                                                              (short)0, acc[nt], false, false);
            acc[nt] = __builtin_amdgcn_wmma_f32_16x16x32_bf16(false, al, false, bh,
                                                              (short)0, acc[nt], false, false);
            if ((nt & 3) == 3) sched_fence();   // bound live B fragments
        }
    }

    // lane l and l^16 each hold half of row (l&15)'s d-range
    xsqp += __shfl_xor(xsqp, 16, 32);
    if (half == 0) s_xsq[wave * 16 + l15] = xsqp;
    __syncthreads();

    float xsqr[8];
    #pragma unroll
    for (int v = 0; v < 8; ++v) xsqr[v] = s_xsq[wave * 16 + 8 * half + v];

    // ---- transform in place: q = rcp(1+max(xsq - 2*cross + csq, 0)); row sums
    float rowsum[8];
    #pragma unroll
    for (int v = 0; v < 8; ++v) rowsum[v] = 0.f;

    #pragma unroll
    for (int nt = 0; nt < 16; ++nt) {
        const float csq = s_csq[nt * 16 + l15];
        #pragma unroll
        for (int v = 0; v < 8; ++v) {
            float s = xsqr[v] + csq;
            float n = fmaxf(fmaf(-2.f, acc[nt][v], s), 0.f);
            float q = fast_rcp(1.f + n);   // ALPHA=1 -> power==1
            acc[nt][v] = q;
            rowsum[v] += q;
        }
    }

    // ---- row-sum reduction across the 16 lanes of each half-wave
    #pragma unroll
    for (int m = 1; m <= 8; m <<= 1) {
        #pragma unroll
        for (int v = 0; v < 8; ++v) rowsum[v] += __shfl_xor(rowsum[v], m, 32);
    }
    float inv[8];
    #pragma unroll
    for (int v = 0; v < 8; ++v) inv[v] = 1.f / rowsum[v];

    // ---- normalize + store (lanes 0-15 -> rows base..base+7, lanes 16-31 -> +8)
    const int rowOut = growBase + 8 * half;
    #pragma unroll
    for (int nt = 0; nt < 16; ++nt) {
        #pragma unroll
        for (int v = 0; v < 8; ++v) {
            out[(rowOut + v) * 256 + nt * 16 + l15] = acc[nt][v] * inv[v];
        }
    }
}

extern "C" void kernel_launch(void* const* d_in, const int* in_sizes, int n_in,
                              void* d_out, int out_size, void* d_ws, size_t ws_size,
                              hipStream_t stream) {
    const float* x  = (const float*)d_in[0];   // [131072, 256] f32
    const float* cc = (const float*)d_in[1];   // [256, 256] f32
    float* out = (float*)d_out;                // [131072, 256] f32
    (void)in_sizes; (void)n_in; (void)out_size; (void)d_ws; (void)ws_size;

    const int N = 131072;
    dim3 grid(N / 128), block(256);
    clusterer_kernel<<<grid, block, 0, stream>>>(x, cc, out);
}
